// MultiHeaded_SelfAttn_36850819399886
// MI455X (gfx1250) — compile-verified, matrix operand order checked by reading, not run on hardware
//
#include <hip/hip_runtime.h>

typedef __attribute__((ext_vector_type(16))) __bf16 v16bf;
typedef __attribute__((ext_vector_type(8)))  __bf16 v8bf;
typedef __attribute__((ext_vector_type(8)))  float  v8f;

#define WMMA_BF16(a, b, c) \
  __builtin_amdgcn_wmma_f32_16x16x32_bf16(false, (a), false, (b), (short)0, (c), false, false)

// LDS byte offset of a __shared__ object (addrspace(3) pointers are the raw
// wave-LDS byte offsets that GLOBAL_LOAD_ASYNC_TO_LDS wants in VDST).
__device__ __forceinline__ unsigned lds_off_u32(const void* p) {
  return (unsigned)(unsigned long long)(__attribute__((address_space(3))) const void*)p;
}

// Per-lane async copy: 16 bytes global -> LDS, tracked by ASYNCcnt.
__device__ __forceinline__ void async_b128(void* lds, const void* g) {
  asm volatile("global_load_async_to_lds_b128 %0, %1, off"
               :: "v"(lds_off_u32(lds)), "v"((unsigned long long)g)
               : "memory");
}

__device__ __forceinline__ void wait_async0() {
  asm volatile("s_wait_asynccnt 0" ::: "memory");
}

// Load a 16x32 bf16 A/B fragment (CDNA5 16-bit WMMA lane layout): caller passes
// p already offset by row*ld + koff (koff = 8*(lane>=16)); elements [0..7] are
// K=koff..koff+7, [8..15] are K=koff+16..koff+23.
__device__ __forceinline__ v16bf ldfrag(const __bf16* p) {
  v8bf lo = *(const v8bf*)(p);
  v8bf hi = *(const v8bf*)(p + 16);
  v16bf f;
#pragma unroll
  for (int i = 0; i < 8; ++i) { f[i] = lo[i]; f[i + 8] = hi[i]; }
  return f;
}

// ---------------------------------------------------------------------------
// Kernel 1: f32 -> bf16 conversion of activations (q and k_and_v, 8M each)
// ---------------------------------------------------------------------------
__global__ void __launch_bounds__(256) k_cvt_act(const float* __restrict__ q,
                                                 const float* __restrict__ kv,
                                                 __bf16* __restrict__ qb,
                                                 __bf16* __restrict__ kvb) {
  size_t i = (size_t)blockIdx.x * 256 + threadIdx.x;
  qb[i]  = (__bf16)q[i];
  kvb[i] = (__bf16)kv[i];
}

// ---------------------------------------------------------------------------
// Kernel 2: transpose+convert weights into B^T [N][K] bf16 layouts.
//   wqT  [1024][1024] : wqT[n][e]  = wq[h,e,d], n = h*64+d
//   wkvT [2048][1024] : rows 0-1023 from wk, rows 1024-2047 from wv
//   woT  [1024][1024] : woT[e][c]  = wo[c][e]
// ---------------------------------------------------------------------------
__global__ void __launch_bounds__(256) k_cvt_w(const float* __restrict__ wq,
                                               const float* __restrict__ wk,
                                               const float* __restrict__ wv,
                                               const float* __restrict__ wo,
                                               __bf16* __restrict__ wqT,
                                               __bf16* __restrict__ wkvT,
                                               __bf16* __restrict__ woT) {
  int i   = blockIdx.x * 256 + threadIdx.x;   // 0 .. 1M-1
  int row = i >> 10;                          // n (for wq/wk/wv) or e (for wo)
  int cc  = i & 1023;                         // e (for wq/wk/wv) or c (for wo)
  int h = row >> 6, d = row & 63;
  size_t src = (size_t)(h * 1024 + cc) * 64 + d;
  wqT[i]                      = (__bf16)wq[src];
  wkvT[i]                     = (__bf16)wk[src];
  wkvT[(size_t)i + 1024*1024] = (__bf16)wv[src];
  woT[i]                      = (__bf16)wo[(size_t)cc * 1024 + row];
}

// ---------------------------------------------------------------------------
// Kernel 3: QKV projection GEMM. M=8192 (b*s), K=1024.
//   mode 0: N=1024 -> qh [bh][s][64] (+bq)
//   mode 1: N=2048, n<1024 -> kh [bh][t][64] (+bk); n>=1024 -> vt [bh][64][t] (+bv)
// 4 waves/WG; 64x64 WG tile; A/B 64-wide K tiles double-buffered in LDS via
// async global->LDS copies.
// ---------------------------------------------------------------------------
__global__ void __launch_bounds__(128) k_gemm_qkv(const __bf16* __restrict__ X,
                                                  const __bf16* __restrict__ WT,
                                                  __bf16* __restrict__ out0,
                                                  __bf16* __restrict__ vt,
                                                  const float* __restrict__ bias0,
                                                  const float* __restrict__ bias1,
                                                  int mode) {
  __shared__ __bf16 als[2][64 * 64];
  __shared__ __bf16 bls[2][64 * 64];

  const int tid = threadIdx.x;
  const int wave = tid >> 5, lane = tid & 31;
  const int col = lane & 15, hi = lane >> 4, koff = hi << 3;
  const int m0 = blockIdx.x * 64 + wave * 16;
  const int n0 = blockIdx.y * 64;

  // async-copy addressing: 128 threads cover a 64x64 bf16 tile in 4 passes of
  // 16 rows; each thread moves 16B (8 elements) per pass.
  const int crow = tid >> 3;        // 0..15
  const int cseg = (tid & 7) << 3;  // element offset within row: 0..56

  const __bf16* xg = X  + (size_t)(blockIdx.x * 64 + crow) * 1024 + cseg;
  const __bf16* wg = WT + (size_t)(n0 + crow) * 1024 + cseg;

#pragma unroll
  for (int p = 0; p < 4; ++p) {
    async_b128(&als[0][(crow + p * 16) * 64 + cseg], xg + (size_t)(p * 16) * 1024);
    async_b128(&bls[0][(crow + p * 16) * 64 + cseg], wg + (size_t)(p * 16) * 1024);
  }

  v8f acc[4] = {};
  for (int i = 0; i < 16; ++i) {
    wait_async0();       // our chunks of tile i have landed
    __syncthreads();     // everyone's chunks landed; iter i-1 reads all done
    if (i + 1 < 16) {
      const int kn = (i + 1) * 64;
#pragma unroll
      for (int p = 0; p < 4; ++p) {
        async_b128(&als[(i + 1) & 1][(crow + p * 16) * 64 + cseg],
                   xg + (size_t)(p * 16) * 1024 + kn);
        async_b128(&bls[(i + 1) & 1][(crow + p * 16) * 64 + cseg],
                   wg + (size_t)(p * 16) * 1024 + kn);
      }
    }
    const __bf16* ab = &als[i & 1][0];
    const __bf16* bb = &bls[i & 1][0];
#pragma unroll
    for (int kk = 0; kk < 64; kk += 32) {
      v16bf a = ldfrag(ab + (wave * 16 + col) * 64 + kk + koff);
#pragma unroll
      for (int j = 0; j < 4; ++j) {
        v16bf b = ldfrag(bb + (j * 16 + col) * 64 + kk + koff);
        acc[j] = WMMA_BF16(a, b, acc[j]);
      }
    }
  }

#pragma unroll
  for (int j = 0; j < 4; ++j) {
    const int n = n0 + j * 16 + col;
#pragma unroll
    for (int r = 0; r < 8; ++r) {
      const int m = m0 + r + (hi << 3);
      const int bb2 = m >> 11, s = m & 2047;
      float v = acc[j][r];
      if (mode == 0 || n < 1024) {
        const int h = n >> 6, d = n & 63;
        out0[(((size_t)(bb2 * 16 + h)) * 2048 + s) * 64 + d] = (__bf16)(v + bias0[n]);
      } else {
        const int nn = n - 1024;
        const int h = nn >> 6, d = nn & 63;
        vt[(((size_t)(bb2 * 16 + h)) * 64 + d) * 2048 + s] = (__bf16)(v + bias1[nn]);
      }
    }
  }
}

// ---------------------------------------------------------------------------
// Kernel 4: flash attention. grid = (64 bh, 32 row-blocks), block = 128.
// K/V 64x64 tiles double-buffered in LDS (async copies, shared by all 4
// waves); per-wave online softmax; P staged through LDS for the PV GEMM.
// ---------------------------------------------------------------------------
__global__ void __launch_bounds__(128) k_attn(const __bf16* __restrict__ qh,
                                              const __bf16* __restrict__ kh,
                                              const __bf16* __restrict__ vt,
                                              __bf16* __restrict__ z) {
  __shared__ __bf16 kls[2][64 * 64];
  __shared__ __bf16 vls[2][64 * 64];
  __shared__ __bf16 pls[4][16 * 64];

  const int tid = threadIdx.x;
  const int wave = tid >> 5, lane = tid & 31;
  const int col = lane & 15, hi = lane >> 4, koff = hi << 3;
  const int bh = blockIdx.x;                   // b*16 + h
  const int s0 = blockIdx.y * 64 + wave * 16;  // query rows of this wave

  const __bf16* qbase = qh + ((size_t)bh * 2048 + s0) * 64;
  const int crow = tid >> 3, cseg = (tid & 7) << 3;
  const __bf16* kg = kh + ((size_t)bh * 2048 + crow) * 64 + cseg;   // +t0*64
  const __bf16* vg = vt + ((size_t)bh * 64 + crow) * 2048 + cseg;   // +t0

#pragma unroll
  for (int p = 0; p < 4; ++p) {
    async_b128(&kls[0][(crow + p * 16) * 64 + cseg], kg + (size_t)(p * 16) * 64);
    async_b128(&vls[0][(crow + p * 16) * 64 + cseg], vg + (size_t)(p * 16) * 2048);
  }

  // Q fragments: rows s0+col, d-dim 64 -> two fragments, kept in regs.
  v16bf aq0 = ldfrag(qbase + (size_t)col * 64 + koff);
  v16bf aq1 = ldfrag(qbase + (size_t)col * 64 + koff + 32);

  v8f O[4] = {};
  float M[8], L[8];
#pragma unroll
  for (int r = 0; r < 8; ++r) { M[r] = -3.0e38f; L[r] = 0.0f; }

  for (int i = 0; i < 32; ++i) {
    wait_async0();
    __syncthreads();
    if (i + 1 < 32) {
      const int tn = (i + 1) * 64;
#pragma unroll
      for (int p = 0; p < 4; ++p) {
        async_b128(&kls[(i + 1) & 1][(crow + p * 16) * 64 + cseg],
                   kg + (size_t)(tn + p * 16) * 64);
        async_b128(&vls[(i + 1) & 1][(crow + p * 16) * 64 + cseg],
                   vg + (size_t)(p * 16) * 2048 + tn);
      }
    }
    const __bf16* kb0 = &kls[i & 1][0];
    const __bf16* vb0 = &vls[i & 1][0];

    // --- scores: 4 column sub-tiles, 2 WMMAs each over d ---
    v8f s[4];
#pragma unroll
    for (int j = 0; j < 4; ++j) {
      const __bf16* kb = kb0 + (j * 16 + col) * 64 + koff;
      v16bf b0 = ldfrag(kb);
      v16bf b1 = ldfrag(kb + 32);
      v8f c = {};
      c = WMMA_BF16(aq0, b0, c);
      c = WMMA_BF16(aq1, b1, c);
      s[j] = c;
    }

    // --- scale + tile row-max (reduce across 16-lane N groups) ---
    float tm[8];
#pragma unroll
    for (int r = 0; r < 8; ++r) tm[r] = -3.0e38f;
#pragma unroll
    for (int j = 0; j < 4; ++j)
#pragma unroll
      for (int r = 0; r < 8; ++r) {
        s[j][r] *= 0.125f;  // 1/sqrt(64)
        tm[r] = fmaxf(tm[r], s[j][r]);
      }
#pragma unroll
    for (int off = 1; off < 16; off <<= 1)
#pragma unroll
      for (int r = 0; r < 8; ++r) tm[r] = fmaxf(tm[r], __shfl_xor(tm[r], off, 32));

    // --- online softmax update ---
    float corr[8];
#pragma unroll
    for (int r = 0; r < 8; ++r) {
      float mn = fmaxf(M[r], tm[r]);
      corr[r] = __expf(M[r] - mn);
      M[r] = mn;
    }
#pragma unroll
    for (int j = 0; j < 4; ++j)
#pragma unroll
      for (int r = 0; r < 8; ++r) O[j][r] *= corr[r];

    float ps[8];
#pragma unroll
    for (int r = 0; r < 8; ++r) ps[r] = 0.0f;
#pragma unroll
    for (int j = 0; j < 4; ++j)
#pragma unroll
      for (int r = 0; r < 8; ++r) {
        float p = __expf(s[j][r] - M[r]);
        ps[r] += p;
        pls[wave][(r + (hi << 3)) * 64 + j * 16 + col] = (__bf16)p;
      }
#pragma unroll
    for (int off = 1; off < 16; off <<= 1)
#pragma unroll
      for (int r = 0; r < 8; ++r) ps[r] += __shfl_xor(ps[r], off, 32);
#pragma unroll
    for (int r = 0; r < 8; ++r) L[r] = L[r] * corr[r] + ps[r];

    __syncthreads();  // P tile visible for fragment reloads

    // --- PV: A from P (LDS), B from V tile (LDS, d-major / t-contiguous) ---
    v16bf ap0 = ldfrag(&pls[wave][col * 64 + koff]);
    v16bf ap1 = ldfrag(&pls[wave][col * 64 + koff + 32]);
#pragma unroll
    for (int j2 = 0; j2 < 4; ++j2) {
      const __bf16* vb = vb0 + (j2 * 16 + col) * 64 + koff;
      v16bf b0 = ldfrag(vb);
      v16bf b1 = ldfrag(vb + 32);
      O[j2] = WMMA_BF16(ap0, b0, O[j2]);
      O[j2] = WMMA_BF16(ap1, b1, O[j2]);
    }
  }

  // --- normalize + store z [b][s][h*64+d] bf16 ---
  const int b = bh >> 4, h = bh & 15;
#pragma unroll
  for (int j2 = 0; j2 < 4; ++j2)
#pragma unroll
    for (int r = 0; r < 8; ++r) {
      const int srow = s0 + r + (hi << 3);
      float o = O[j2][r] / L[r];
      z[((size_t)(b * 2048 + srow)) * 1024 + h * 64 + j2 * 16 + col] = (__bf16)o;
    }
}

// ---------------------------------------------------------------------------
// Kernel 5: output projection GEMM. M=8192, K=1024, N=1024; f32 out + bo.
// Same LDS double-buffered async staging as kernel 3.
// ---------------------------------------------------------------------------
__global__ void __launch_bounds__(128) k_gemm_out(const __bf16* __restrict__ Z,
                                                  const __bf16* __restrict__ WoT,
                                                  const float* __restrict__ bo,
                                                  float* __restrict__ out) {
  __shared__ __bf16 als[2][64 * 64];
  __shared__ __bf16 bls[2][64 * 64];

  const int tid = threadIdx.x;
  const int wave = tid >> 5, lane = tid & 31;
  const int col = lane & 15, hi = lane >> 4, koff = hi << 3;
  const int m0 = blockIdx.x * 64 + wave * 16;
  const int n0 = blockIdx.y * 64;

  const int crow = tid >> 3, cseg = (tid & 7) << 3;
  const __bf16* zg = Z   + (size_t)(blockIdx.x * 64 + crow) * 1024 + cseg;
  const __bf16* wg = WoT + (size_t)(n0 + crow) * 1024 + cseg;

#pragma unroll
  for (int p = 0; p < 4; ++p) {
    async_b128(&als[0][(crow + p * 16) * 64 + cseg], zg + (size_t)(p * 16) * 1024);
    async_b128(&bls[0][(crow + p * 16) * 64 + cseg], wg + (size_t)(p * 16) * 1024);
  }

  v8f acc[4] = {};
  for (int i = 0; i < 16; ++i) {
    wait_async0();
    __syncthreads();
    if (i + 1 < 16) {
      const int kn = (i + 1) * 64;
#pragma unroll
      for (int p = 0; p < 4; ++p) {
        async_b128(&als[(i + 1) & 1][(crow + p * 16) * 64 + cseg],
                   zg + (size_t)(p * 16) * 1024 + kn);
        async_b128(&bls[(i + 1) & 1][(crow + p * 16) * 64 + cseg],
                   wg + (size_t)(p * 16) * 1024 + kn);
      }
    }
    const __bf16* ab = &als[i & 1][0];
    const __bf16* bb = &bls[i & 1][0];
#pragma unroll
    for (int kk = 0; kk < 64; kk += 32) {
      v16bf a = ldfrag(ab + (wave * 16 + col) * 64 + kk + koff);
#pragma unroll
      for (int j = 0; j < 4; ++j) {
        v16bf b = ldfrag(bb + (j * 16 + col) * 64 + kk + koff);
        acc[j] = WMMA_BF16(a, b, acc[j]);
      }
    }
  }

#pragma unroll
  for (int j = 0; j < 4; ++j) {
    const int n = n0 + j * 16 + col;
#pragma unroll
    for (int r = 0; r < 8; ++r) {
      const int m = m0 + r + (hi << 3);
      out[(size_t)m * 1024 + n] = acc[j][r] + bo[n];
    }
  }
}

// ---------------------------------------------------------------------------
// Host launcher
// ---------------------------------------------------------------------------
extern "C" void kernel_launch(void* const* d_in, const int* in_sizes, int n_in,
                              void* d_out, int out_size, void* d_ws, size_t ws_size,
                              hipStream_t stream) {
  const float* q  = (const float*)d_in[0];
  const float* kv = (const float*)d_in[1];
  const float* wq = (const float*)d_in[2];
  const float* bq = (const float*)d_in[3];
  const float* wk = (const float*)d_in[4];
  const float* bk = (const float*)d_in[5];
  const float* wv = (const float*)d_in[6];
  const float* bv = (const float*)d_in[7];
  const float* wo = (const float*)d_in[8];
  const float* bo = (const float*)d_in[9];
  float* out = (float*)d_out;

  char* w = (char*)d_ws;
  const size_t MB = 1024 * 1024;
  __bf16* qbf  = (__bf16*)(w + 0 * MB);    // 16 MB (reused later for z)
  __bf16* kvbf = (__bf16*)(w + 16 * MB);   // 16 MB
  __bf16* wqT  = (__bf16*)(w + 32 * MB);   //  2 MB
  __bf16* wkvT = (__bf16*)(w + 34 * MB);   //  4 MB
  __bf16* woT  = (__bf16*)(w + 38 * MB);   //  2 MB
  __bf16* qhB  = (__bf16*)(w + 40 * MB);   // 16 MB
  __bf16* khB  = (__bf16*)(w + 56 * MB);   // 16 MB
  __bf16* vtB  = (__bf16*)(w + 72 * MB);   // 16 MB  (total 88 MB)
  __bf16* zB   = qbf;                      // reuse q_bf region after projection

  k_cvt_act<<<dim3(8388608 / 256), dim3(256), 0, stream>>>(q, kv, qbf, kvbf);
  k_cvt_w<<<dim3(1048576 / 256), dim3(256), 0, stream>>>(wq, wk, wv, wo, wqT, wkvT, woT);
  k_gemm_qkv<<<dim3(128, 16), dim3(128), 0, stream>>>(qbf, wqT, qhB, nullptr, bq, nullptr, 0);
  k_gemm_qkv<<<dim3(128, 32), dim3(128), 0, stream>>>(kvbf, wkvT, khB, vtB, bk, bv, 1);
  k_attn<<<dim3(64, 32), dim3(128), 0, stream>>>(qhB, khB, vtB, zB);
  k_gemm_out<<<dim3(128, 16), dim3(128), 0, stream>>>(zB, woT, bo, out);
}
